// DeformableTransformerEncoderLayer_10136122818999
// MI455X (gfx1250) — matrix operand order repaired and checked by reading, not compiled, not run on hardware
//
#include <hip/hip_runtime.h>
#include <hip/hip_bf16.h>

typedef __bf16 bf16;
typedef __attribute__((ext_vector_type(16))) __bf16 v16bf;
typedef __attribute__((ext_vector_type(8)))  __bf16 v8bf;
typedef __attribute__((ext_vector_type(8)))  float  v8f;

#define BS_  2
#define TT_  4
#define HS_  32
#define WSP_ 32
#define HW_  (HS_ * WSP_)      // 1024
#define D_   256
#define NH_  8
#define NL_  4
#define NP_  4
#define HD_  (D_ / NH_)        // 32
#define DFFN_ 1024
#define LQ_  (TT_ * HW_)       // 4096
#define M_   (BS_ * LQ_)       // 8192

// ---------------------------------------------------------------------------
// Pack a row-major fp32 weight matrix W[K][N] into WMMA B-fragment order:
// fragment f = kt*(N/16)+nt holds 32 lanes x 16 bf16 contiguous, where for
// lane l: col n = nt*16 + (l&15), element e: k = kt*32 + 16*(l>>4) + e.
// ---------------------------------------------------------------------------
__global__ void pack_b_kernel(const float* __restrict__ W, bf16* __restrict__ P,
                              int K, int N) {
    int t = blockIdx.x * blockDim.x + threadIdx.x;
    int ntiles = N >> 4;
    int total = (K >> 5) * ntiles * 32;
    if (t >= total) return;
    int lane = t & 31;
    int f    = t >> 5;
    int nt   = f % ntiles;
    int kt   = f / ntiles;
    int n     = nt * 16 + (lane & 15);
    int kbase = kt * 32 + ((lane >> 4) << 4);
    v16bf frag;
#pragma unroll
    for (int e = 0; e < 16; ++e)
        frag[e] = (bf16)W[(size_t)(kbase + e) * N + n];
    *(v16bf*)(P + (size_t)t * 16) = frag;
}

__global__ void f32_to_bf16_kernel(const float* __restrict__ X,
                                   bf16* __restrict__ Y, int n) {
    int i = blockIdx.x * blockDim.x + threadIdx.x;
    if (i < n) Y[i] = (bf16)X[i];
}

// ---------------------------------------------------------------------------
// bf16 WMMA GEMM, 16x64 output tile per wave32 (4 accumulators, A reused
// across 4 B fragments per K-step of 32). Optional ReLU; optional fp32/bf16
// stores. v_wmma_f32_16x16x32_bf16 with fp32 accumulation.
// ---------------------------------------------------------------------------
__global__ void gemm_bf16_wmma4(const bf16* __restrict__ A,
                                const bf16* __restrict__ Bp,
                                const float* __restrict__ bias,
                                float* __restrict__ Cf, bf16* __restrict__ Cb,
                                int M, int N, int K, int relu) {
    const int lane    = threadIdx.x & 31;
    const int wv      = threadIdx.x >> 5;
    const int ntiles  = N >> 4;        // 16-col tiles
    const int ngroups = N >> 6;        // 64-col groups
    const int gw      = blockIdx.x * (blockDim.x >> 5) + wv;
    const int mt      = gw / ngroups;
    const int ng      = gw - mt * ngroups;
    if (mt * 16 >= M) return;

    const int half = lane >> 4;
    const int lid  = lane & 15;
    const long row = (long)(mt * 16 + lid);
    const int  nt0 = ng * 4;           // first 16-col tile of this group

    v8f acc0 = {0.f,0.f,0.f,0.f,0.f,0.f,0.f,0.f};
    v8f acc1 = acc0, acc2 = acc0, acc3 = acc0;

    const int nk = K >> 5;
    const bf16* arow = A + row * (long)K + half * 16;

    for (int kt = 0; kt < nk; ++kt) {
        // A fragment (contiguous k = 16*half + e, matching packed B): two aligned b128 loads
        union { v16bf v; v8bf h[2]; } af;
        const bf16* ab = arow + kt * 32;
        af.h[0] = *(const v8bf*)(ab);
        af.h[1] = *(const v8bf*)(ab + 8);
        if (kt + 1 < nk) __builtin_prefetch(arow + (kt + 1) * 32, 0, 3);

        // 4 B fragments: contiguous 32B per lane each, one A reuse group
        const bf16* bb = Bp + ((size_t)(kt * ntiles + nt0) * 32 + lane) * 16;
        v16bf b0 = *(const v16bf*)(bb);
        v16bf b1 = *(const v16bf*)(bb + 512);
        v16bf b2 = *(const v16bf*)(bb + 1024);
        v16bf b3 = *(const v16bf*)(bb + 1536);

        acc0 = __builtin_amdgcn_wmma_f32_16x16x32_bf16(
            false, af.v, false, b0, (short)0, acc0, false, false);
        acc1 = __builtin_amdgcn_wmma_f32_16x16x32_bf16(
            false, af.v, false, b1, (short)0, acc1, false, false);
        acc2 = __builtin_amdgcn_wmma_f32_16x16x32_bf16(
            false, af.v, false, b2, (short)0, acc2, false, false);
        acc3 = __builtin_amdgcn_wmma_f32_16x16x32_bf16(
            false, af.v, false, b3, (short)0, acc3, false, false);
    }

    v8f accs[4] = {acc0, acc1, acc2, acc3};
#pragma unroll
    for (int j = 0; j < 4; ++j) {
        const int col = (nt0 + j) * 16 + lid;
        float bv = bias ? bias[col] : 0.0f;
#pragma unroll
        for (int r = 0; r < 8; ++r) {
            long m  = (long)mt * 16 + half * 8 + r;  // C/D layout: VGPR r -> M = r + 8*half
            float v = accs[j][r] + bv;
            if (relu) v = fmaxf(v, 0.0f);
            if (Cf) Cf[m * N + col] = v;
            if (Cb) Cb[m * N + col] = (bf16)v;
        }
    }
}

// Softmax over the 16 (level,point) logits per (token, head), in place.
__global__ void softmax16_kernel(float* __restrict__ L) {
    int i = blockIdx.x * blockDim.x + threadIdx.x;   // i = m*NH + h
    if (i >= M_ * NH_) return;
    float* p = L + (size_t)i * 16;
    float v[16], mx = -1e30f;
#pragma unroll
    for (int e = 0; e < 16; ++e) { v[e] = p[e]; mx = fmaxf(mx, v[e]); }
    float s = 0.f;
#pragma unroll
    for (int e = 0; e < 16; ++e) { v[e] = expf(v[e] - mx); s += v[e]; }
    float inv = 1.0f / s;
#pragma unroll
    for (int e = 0; e < 16; ++e) p[e] = v[e] * inv;
}

// sampling_locations = reference_points + offsets / (W,H); written to d_out.
__global__ void samp_loc_kernel(const float* __restrict__ offs,
                                const float* __restrict__ ref,
                                const int* __restrict__ ss,
                                float* __restrict__ sloc) {
    int i = blockIdx.x * blockDim.x + threadIdx.x;   // ((m*NH+h)*NL+l)*NP+p
    if (i >= M_ * NH_ * NL_ * NP_) return;
    int p = i & 3;
    int l = (i >> 2) & 3;
    int h = (i >> 4) & 7;
    int m = i >> 7;
    float Wl = (float)ss[l * 2 + 1];
    float Hl = (float)ss[l * 2 + 0];
    size_t oc = (size_t)m * D_ + ((((h * NL_ + l) * NP_ + p)) << 1);
    const float* r = ref + (((size_t)m * NL_ + l) << 1);
    sloc[(size_t)i * 2 + 0] = r[0] + offs[oc + 0] / Wl;
    sloc[(size_t)i * 2 + 1] = r[1] + offs[oc + 1] / Hl;
}

// Deformable attention: one wave per (token m, head h); lane = channel (HD==32).
// Every bilinear tap is a coalesced 128B load across the wave.
__global__ void deform_attn_kernel(const float* __restrict__ value,
                                   const float* __restrict__ sloc,
                                   const float* __restrict__ aw,
                                   const int* __restrict__ ss,
                                   const int* __restrict__ lsi,
                                   float* __restrict__ outF,
                                   bf16* __restrict__ outB) {
    int gw = blockIdx.x * (blockDim.x >> 5) + (threadIdx.x >> 5);
    if (gw >= M_ * NH_) return;
    int lane = threadIdx.x & 31;
    int h = gw & (NH_ - 1);
    int m = gw >> 3;
    int b = m >> 12;                      // LQ_ = 4096 = 2^12
    float acc = 0.f;
#pragma unroll
    for (int l = 0; l < NL_; ++l) {
        int Hl = ss[l * 2], Wl = ss[l * 2 + 1];
        int start = lsi[l];
        const float* vbase = value + ((size_t)b * LQ_ + start) * D_ + h * HD_;
#pragma unroll
        for (int p = 0; p < NP_; ++p) {
            size_t si = (((size_t)gw * NL_ + l) * NP_ + p) * 2;
            float lx = sloc[si + 0];
            float ly = sloc[si + 1];
            float w  = aw[(size_t)gw * 16 + l * NP_ + p];
            float x = lx * (float)Wl - 0.5f;
            float y = ly * (float)Hl - 0.5f;
            float x0 = floorf(x), y0 = floorf(y);
            float fx = x - x0, fy = y - y0;
            int ix = (int)x0, iy = (int)y0;
#pragma unroll
            for (int dy = 0; dy < 2; ++dy) {
#pragma unroll
                for (int dx = 0; dx < 2; ++dx) {
                    int xi = ix + dx, yi = iy + dy;
                    float wt = (dx ? fx : 1.f - fx) * (dy ? fy : 1.f - fy);
                    if (xi >= 0 && xi < Wl && yi >= 0 && yi < Hl) {
                        float g = vbase[((size_t)(yi * Wl + xi)) * D_ + lane];
                        acc += w * wt * g;
                    }
                }
            }
        }
    }
    size_t o = (size_t)m * D_ + h * HD_ + lane;
    outF[o] = acc;
    outB[o] = (bf16)acc;
}

// Residual add + LayerNorm; one 256-thread block per row (D_ == 256).
__global__ void add_norm_kernel(const float* __restrict__ X,
                                const float* __restrict__ Y,
                                const float* __restrict__ g,
                                const float* __restrict__ bta,
                                float* __restrict__ outF,
                                bf16* __restrict__ outB) {
    __shared__ float s1[256], s2[256];
    int row = blockIdx.x;
    int t = threadIdx.x;
    size_t idx = (size_t)row * D_ + t;
    float v = X[idx] + Y[idx];
    s1[t] = v; s2[t] = v * v;
    __syncthreads();
    for (int s = 128; s > 0; s >>= 1) {
        if (t < s) { s1[t] += s1[t + s]; s2[t] += s2[t + s]; }
        __syncthreads();
    }
    float mean = s1[0] * (1.0f / D_);
    float var  = s2[0] * (1.0f / D_) - mean * mean;
    float inv  = rsqrtf(var + 1e-5f);
    float o = (v - mean) * inv * g[t] + bta[t];
    outF[idx] = o;
    if (outB) outB[idx] = (bf16)o;
}

// ---------------------------------------------------------------------------
extern "C" void kernel_launch(void* const* d_in, const int* in_sizes, int n_in,
                              void* d_out, int out_size, void* d_ws, size_t ws_size,
                              hipStream_t stream) {
    (void)in_sizes; (void)n_in; (void)out_size; (void)ws_size;
    const float* src    = (const float*)d_in[0];
    const float* ref    = (const float*)d_in[2];
    const float* w_val  = (const float*)d_in[3];
    const float* b_val  = (const float*)d_in[4];
    const float* w_off  = (const float*)d_in[5];
    const float* b_off  = (const float*)d_in[6];
    const float* w_attn = (const float*)d_in[7];
    const float* b_attn = (const float*)d_in[8];
    const float* w_out  = (const float*)d_in[9];
    const float* b_out  = (const float*)d_in[10];
    const float* g1     = (const float*)d_in[11];
    const float* bb1    = (const float*)d_in[12];
    const float* w_l1   = (const float*)d_in[13];
    const float* b_l1   = (const float*)d_in[14];
    const float* w_l2   = (const float*)d_in[15];
    const float* b_l2   = (const float*)d_in[16];
    const float* g2     = (const float*)d_in[17];
    const float* bb2    = (const float*)d_in[18];
    const int*   ss     = (const int*)d_in[19];
    const int*   lsi    = (const int*)d_in[20];

    char* ws = (char*)d_ws;
    size_t off = 0;
    auto alloc = [&](size_t bytes) -> void* {
        void* p = ws + off;
        off += (bytes + 255) & ~(size_t)255;
        return p;
    };

    // packed bf16 weights
    bf16* pw_val  = (bf16*)alloc((size_t)256 * 256 * 2);
    bf16* pw_off  = (bf16*)alloc((size_t)256 * 256 * 2);
    bf16* pw_attn = (bf16*)alloc((size_t)256 * 128 * 2);
    bf16* pw_out  = (bf16*)alloc((size_t)256 * 256 * 2);
    bf16* pw_l1   = (bf16*)alloc((size_t)256 * 1024 * 2);
    bf16* pw_l2   = (bf16*)alloc((size_t)1024 * 256 * 2);
    // activations / intermediates
    bf16*  xb       = (bf16*) alloc((size_t)M_ * D_ * 2);
    float* value    = (float*)alloc((size_t)M_ * D_ * 4);
    float* offsets  = (float*)alloc((size_t)M_ * D_ * 4);
    float* aw       = (float*)alloc((size_t)M_ * (NH_ * NL_ * NP_) * 4);
    float* attn_f   = (float*)alloc((size_t)M_ * D_ * 4);
    bf16*  attn_b   = (bf16*) alloc((size_t)M_ * D_ * 2);
    float* proj     = (float*)alloc((size_t)M_ * D_ * 4);
    float* x1f      = (float*)alloc((size_t)M_ * D_ * 4);
    bf16*  x1b      = (bf16*) alloc((size_t)M_ * D_ * 2);
    bf16*  hidden_b = (bf16*) alloc((size_t)M_ * DFFN_ * 2);
    float* ffn_out  = (float*)alloc((size_t)M_ * D_ * 4);

    float* xout = (float*)d_out;                 // (BS,T,HW,D)
    float* sloc = xout + (size_t)M_ * D_;        // (bs,Lq,NH,NL,NP,2)

    auto pack = [&](const float* W, bf16* P, int K, int N) {
        int threads = (K >> 5) * (N >> 4) * 32;
        pack_b_kernel<<<dim3((threads + 255) / 256), dim3(256), 0, stream>>>(W, P, K, N);
    };
    auto gemm = [&](const bf16* A, const bf16* Bp, const float* bias,
                    float* Cf, bf16* Cb, int M, int N, int K, int relu) {
        int waves = (M / 16) * (N / 64);
        gemm_bf16_wmma4<<<dim3((waves + 7) / 8), dim3(256), 0, stream>>>(
            A, Bp, bias, Cf, Cb, M, N, K, relu);
    };

    // 1) weight packing (fp32 -> WMMA-fragment bf16)
    pack(w_val,  pw_val,  256, 256);
    pack(w_off,  pw_off,  256, 256);
    pack(w_attn, pw_attn, 256, 128);
    pack(w_out,  pw_out,  256, 256);
    pack(w_l1,   pw_l1,   256, 1024);
    pack(w_l2,   pw_l2,   1024, 256);

    // 2) x -> bf16 (GEMM A operand)
    {
        int n = M_ * D_;
        f32_to_bf16_kernel<<<dim3((n + 255) / 256), dim3(256), 0, stream>>>(src, xb, n);
    }

    // 3) projections (WMMA)
    gemm(xb, pw_val,  b_val,  value,   nullptr, M_, 256, 256, 0);
    gemm(xb, pw_off,  b_off,  offsets, nullptr, M_, 256, 256, 0);
    gemm(xb, pw_attn, b_attn, aw,      nullptr, M_, 128, 256, 0);

    // 4) softmax over 16 per (token, head), in place
    {
        int n = M_ * NH_;
        softmax16_kernel<<<dim3((n + 255) / 256), dim3(256), 0, stream>>>(aw);
    }

    // 5) sampling locations -> second output
    {
        int n = M_ * NH_ * NL_ * NP_;
        samp_loc_kernel<<<dim3((n + 255) / 256), dim3(256), 0, stream>>>(offsets, ref, ss, sloc);
    }

    // 6) deformable attention (wave per token-head, lane = channel)
    {
        int waves = M_ * NH_;
        deform_attn_kernel<<<dim3(waves / 8), dim3(256), 0, stream>>>(
            value, sloc, aw, ss, lsi, attn_f, attn_b);
    }

    // 7) output projection (WMMA)
    gemm(attn_b, pw_out, b_out, proj, nullptr, M_, 256, 256, 0);

    // 8) residual + LN1 (fused bf16 copy for FFN A)
    add_norm_kernel<<<dim3(M_), dim3(256), 0, stream>>>(src, proj, g1, bb1, x1f, x1b);

    // 9) FFN (WMMA): up + ReLU (bf16-only out), then down
    gemm(x1b, pw_l1, b_l1, nullptr, hidden_b, M_, 1024, 256, 1);
    gemm(hidden_b, pw_l2, b_l2, ffn_out, nullptr, M_, 256, 1024, 0);

    // 10) residual + LN2 -> first output
    add_norm_kernel<<<dim3(M_), dim3(256), 0, stream>>>(x1f, ffn_out, g2, bb2, xout, nullptr);
}
